// EncoderBlock_40827959116462
// MI455X (gfx1250) — matrix operand order repaired and checked by reading, not compiled, not run on hardware
//
#include <hip/hip_runtime.h>
#include <hip/hip_bf16.h>
#include <math.h>
#include <stdint.h>

typedef __attribute__((ext_vector_type(8)))  __bf16 v8bf;
typedef __attribute__((ext_vector_type(16))) __bf16 v16bf;
typedef __attribute__((ext_vector_type(8)))  float  v8f;
typedef __attribute__((address_space(3))) __bf16 lds_bf16;

#define MTOT 4096   // B*S
#define DM   1024
#define DFF  4096
#define NH   16
#define DK   64
#define SEQ  2048

static __device__ inline v8f wmma_bf16(v16bf a, v16bf b, v8f c) {
  return __builtin_amdgcn_wmma_f32_16x16x32_bf16(false, a, false, b, (short)0, c, false, false);
}

static __device__ inline v16bf cat8(v8bf lo, v8bf hi) {
  v16bf r;
#pragma unroll
  for (int i = 0; i < 8; ++i) { r[i] = lo[i]; r[8 + i] = hi[i]; }
  return r;
}

// issue a 32-byte per-lane async copy global -> LDS (two B128 ops, ASYNCcnt += 2)
static __device__ inline void async_copy32(uint32_t lds_off, const void* gptr) {
  uint64_t ga = (uint64_t)(uintptr_t)gptr;
  asm volatile("global_load_async_to_lds_b128 %0, %1, off\n\t"
               "global_load_async_to_lds_b128 %0, %1, off offset:16"
               :
               : "v"(lds_off), "v"(ga)
               : "memory");
}
static __device__ inline void wait_async0() {
  asm volatile("s_wait_asynccnt 0x0" ::: "memory");
}

// ---------------- LayerNorm (f32 in -> bf16 out), one row per block -------------
__global__ void __launch_bounds__(256) ln_bf16_kernel(const float* __restrict__ x,
                                                      const float* __restrict__ g,
                                                      const float* __restrict__ bta,
                                                      __bf16* __restrict__ out) {
  const int row = blockIdx.x;
  const int tid = threadIdx.x;
  const float* xr = x + (size_t)row * DM;
  float v[4];
  float s = 0.f;
#pragma unroll
  for (int i = 0; i < 4; ++i) { v[i] = xr[tid + i * 256]; s += v[i]; }
  __shared__ float red[256];
  red[tid] = s;
  __syncthreads();
  for (int o = 128; o > 0; o >>= 1) {
    if (tid < o) red[tid] += red[tid + o];
    __syncthreads();
  }
  const float mean = red[0] * (1.0f / DM);
  __syncthreads();
  float s2 = 0.f;
#pragma unroll
  for (int i = 0; i < 4; ++i) { float d = v[i] - mean; s2 += d * d; }
  red[tid] = s2;
  __syncthreads();
  for (int o = 128; o > 0; o >>= 1) {
    if (tid < o) red[tid] += red[tid + o];
    __syncthreads();
  }
  const float inv = rsqrtf(red[0] * (1.0f / DM) + 1e-6f);
#pragma unroll
  for (int i = 0; i < 4; ++i) {
    int c = tid + i * 256;
    out[(size_t)row * DM + c] = (__bf16)(g[c] * (v[i] - mean) * inv + bta[c]);
  }
}

// -------- Transpose f32 W[R][C] -> bf16 Wt[C][R]  (B operand: Wt[k][n]) --------
__global__ void __launch_bounds__(256) transpose_bf16_kernel(const float* __restrict__ W,
                                                             __bf16* __restrict__ Wt,
                                                             int R, int C) {
  size_t idx = (size_t)blockIdx.x * 256 + threadIdx.x;
  if (idx >= (size_t)R * C) return;
  int r = (int)(idx / C);
  int c = (int)(idx % C);
  Wt[(size_t)c * R + r] = (__bf16)W[idx];
}

// ---------------- WMMA GEMM: C[M][N] = A[M][K] * Bt[K][N] + bias ---------------
// Block tile 128x128, 8 waves. B tile (32x128) double-buffered in LDS via
// global_load_async_to_lds_b128 (ASYNCcnt); A direct from global.
// MODE 0: bf16 row-major   MODE 1: bf16 transposed (out[n][m])
// MODE 2: f32 out = resid + C   MODE 3: bf16 gelu(C)
template <int MODE>
__global__ void __launch_bounds__(256) gemm_wmma_kernel(const __bf16* __restrict__ A,
                                                        const __bf16* __restrict__ Bt,
                                                        const float* __restrict__ bias,
                                                        const float* __restrict__ resid,
                                                        void* __restrict__ outp,
                                                        int M, int N, int K) {
  const int tid  = threadIdx.x;
  const int lane = tid & 31;
  const int wave = tid >> 5;
  const int bm = blockIdx.y * 128;
  const int bn = blockIdx.x * 128;
  const int mrow = bm + wave * 16 + (lane & 15);
  const int kbA = (lane < 16) ? 0 : 8;   // A-layout K base per lane half

  __shared__ alignas(32) __bf16 bsh[2][32][136];   // double-buffered 32x128 (+8 pad)

  // cooperative staging map: 256 threads cover 32 rows x 8 chunks of 16 elems
  const int sr = tid >> 3;
  const int sc = (tid & 7) * 16;
  const uint32_t lds_off0 = (uint32_t)(uintptr_t)(lds_bf16*)&bsh[0][sr][sc];
  const uint32_t lds_off1 = (uint32_t)(uintptr_t)(lds_bf16*)&bsh[1][sr][sc];

  v8f acc[8] = {};

  // prologue: async-stage first B tile into buffer 0
  async_copy32(lds_off0, Bt + (size_t)sr * N + bn + sc);

  for (int k = 0; k < K; k += 32) {
    const int buf = (k >> 5) & 1;
    wait_async0();                         // this wave's tile copy done
    __syncthreads();                       // whole block's tile copy visible

    if (k + 32 < K) {                      // async-stage next tile, other buffer
      async_copy32(buf ? lds_off0 : lds_off1,
                   Bt + (size_t)(k + 32 + sr) * N + bn + sc);
    }
    if (k + 64 < K) {                      // gfx1250 global_prefetch_b8
      __builtin_prefetch(Bt + (size_t)(k + 64 + sr) * N + bn + sc, 0, 0);
    }

    const __bf16* arow = A + (size_t)mrow * K + k + kbA;
    v16bf a = cat8(*(const v8bf*)arow, *(const v8bf*)(arow + 16));
#pragma unroll
    for (int t = 0; t < 8; ++t) {
      v16bf b;
#pragma unroll
      for (int i = 0; i < 16; ++i) b[i] = bsh[buf][kbA + (i < 8 ? i : 8 + i)][t * 16 + (lane & 15)];
      acc[t] = wmma_bf16(a, b, acc[t]);
    }
  }

  const int lh = lane >> 4, nlo = lane & 15;
#pragma unroll
  for (int t = 0; t < 8; ++t) {
#pragma unroll
    for (int v = 0; v < 8; ++v) {
      const int m = bm + wave * 16 + v + 8 * lh;  // C/D layout: row = v + 8*(lane>=16)
      const int n = bn + t * 16 + nlo;            // col = lane&15
      float c = acc[t][v] + bias[n];
      if (MODE == 2) {
        ((float*)outp)[(size_t)m * N + n] = resid[(size_t)m * N + n] + c;
      } else if (MODE == 1) {
        ((__bf16*)outp)[(size_t)n * M + m] = (__bf16)c;
      } else if (MODE == 3) {
        float gel = 0.5f * c * (1.0f + erff(c * 0.70710678118654752f));
        ((__bf16*)outp)[(size_t)m * N + n] = (__bf16)gel;
      } else {
        ((__bf16*)outp)[(size_t)m * N + n] = (__bf16)c;
      }
    }
  }
}

// -------- Flash attention: Q[m][1024], Kt[1024][4096], V[m][1024] -> ctx bf16 ---
// 4 waves per block; each wave owns one (batch, head, 16-query tile).
__global__ void __launch_bounds__(128) flash_attn_kernel(const __bf16* __restrict__ Q,
                                                         const __bf16* __restrict__ Kt,
                                                         const __bf16* __restrict__ V,
                                                         __bf16* __restrict__ ctx) {
  const int lane = threadIdx.x & 31;
  const int wave = threadIdx.x >> 5;
  const int gt = blockIdx.x * 4 + wave;          // global tile id
  const int qt = gt & 127;                        // query tile in sequence
  const int h  = (gt >> 7) & (NH - 1);
  const int bb = gt >> 11;                        // batch
  const int lh = lane >> 4, nlo = lane & 15;
  const int kb = (lane < 16) ? 0 : 8;

  // Load Q tile as two A-fragments (K = d_k in steps of 32)
  const int qrow = bb * SEQ + qt * 16 + nlo;      // A lanes: m = lane & 15
  v16bf aq[2];
#pragma unroll
  for (int ks = 0; ks < 2; ++ks) {
    const __bf16* qp = Q + (size_t)qrow * DM + h * DK + ks * 32 + kb;
    aq[ks] = cat8(*(const v8bf*)qp, *(const v8bf*)(qp + 16));
  }

  float mrun[8], lrun[8];
#pragma unroll
  for (int v = 0; v < 8; ++v) { mrun[v] = -1e30f; lrun[v] = 0.f; }
  v8f co[4] = {};

  __shared__ alignas(64) __bf16 pls[4][16 * 32];
  __bf16* myp = pls[wave];

  for (int kt = 0; kt < SEQ; kt += 32) {
    // scores: two 16x16 tiles over 32 keys, K-dim = d_k = 64 (2 wmma each)
    v8f s0 = {}, s1 = {};
#pragma unroll
    for (int ks = 0; ks < 2; ++ks) {
      v16bf bk0, bk1;
#pragma unroll
      for (int i = 0; i < 16; ++i) {
        const __bf16* kcol = Kt + (size_t)(h * DK + ks * 32 + kb + (i < 8 ? i : 8 + i)) * MTOT + bb * SEQ + kt + nlo;
        bk0[i] = kcol[0];
        bk1[i] = kcol[16];
      }
      s0 = wmma_bf16(aq[ks], bk0, s0);
      s1 = wmma_bf16(aq[ks], bk1, s1);
    }
    // online softmax update (rows replicated across each 16-lane group)
#pragma unroll
    for (int v = 0; v < 8; ++v) {
      float x0 = s0[v] * 0.125f, x1 = s1[v] * 0.125f;   // 1/sqrt(64)
      float tm = fmaxf(x0, x1);
      tm = fmaxf(tm, __shfl_xor(tm, 8, 32));
      tm = fmaxf(tm, __shfl_xor(tm, 4, 32));
      tm = fmaxf(tm, __shfl_xor(tm, 2, 32));
      tm = fmaxf(tm, __shfl_xor(tm, 1, 32));
      float mn = fmaxf(mrun[v], tm);
      float sf = __expf(mrun[v] - mn);
      float p0 = __expf(x0 - mn), p1 = __expf(x1 - mn);
      float rs = p0 + p1;
      rs += __shfl_xor(rs, 8, 32);
      rs += __shfl_xor(rs, 4, 32);
      rs += __shfl_xor(rs, 2, 32);
      rs += __shfl_xor(rs, 1, 32);
      lrun[v] = lrun[v] * sf + rs;
      mrun[v] = mn;
#pragma unroll
      for (int t = 0; t < 4; ++t) co[t][v] *= sf;
      myp[(v + 8 * lh) * 32 + nlo]      = (__bf16)p0;
      myp[(v + 8 * lh) * 32 + 16 + nlo] = (__bf16)p1;
    }
    __syncthreads();
    // reload P tile in A-matrix layout
    const int m = nlo;
    v16bf ap = cat8(*(const v8bf*)&myp[m * 32 + kb],
                    *(const v8bf*)&myp[m * 32 + 16 + kb]);
#pragma unroll
    for (int t = 0; t < 4; ++t) {
      v16bf bv;
#pragma unroll
      for (int i = 0; i < 16; ++i)
        bv[i] = V[(size_t)(bb * SEQ + kt + kb + (i < 8 ? i : 8 + i)) * DM + h * DK + t * 16 + nlo];
      co[t] = wmma_bf16(ap, bv, co[t]);
    }
    __syncthreads();
  }

#pragma unroll
  for (int t = 0; t < 4; ++t) {
#pragma unroll
    for (int v = 0; v < 8; ++v) {
      float val = co[t][v] / lrun[v];
      ctx[(size_t)(bb * SEQ + qt * 16 + v + 8 * lh) * DM + h * DK + t * 16 + nlo] =
          (__bf16)val;
    }
  }
}

extern "C" void kernel_launch(void* const* d_in, const int* in_sizes, int n_in,
                              void* d_out, int out_size, void* d_ws, size_t ws_size,
                              hipStream_t stream) {
  const float* x   = (const float*)d_in[0];
  const float* Wq  = (const float*)d_in[1];
  const float* bq  = (const float*)d_in[2];
  const float* Wk  = (const float*)d_in[3];
  const float* bk  = (const float*)d_in[4];
  const float* Wv  = (const float*)d_in[5];
  const float* bv  = (const float*)d_in[6];
  const float* Wo  = (const float*)d_in[7];
  const float* bo  = (const float*)d_in[8];
  const float* W1  = (const float*)d_in[9];
  const float* b1  = (const float*)d_in[10];
  const float* W2  = (const float*)d_in[11];
  const float* b2  = (const float*)d_in[12];
  const float* g1  = (const float*)d_in[13];
  const float* be1 = (const float*)d_in[14];
  const float* g2  = (const float*)d_in[15];
  const float* be2 = (const float*)d_in[16];
  float* out = (float*)d_out;

  char* p = (char*)d_ws;
  auto carve = [&](size_t bytes) -> void* {
    void* r = (void*)p;
    p += (bytes + 255) & ~(size_t)255;
    return r;
  };
  __bf16* h_bf   = (__bf16*)carve((size_t)MTOT * DM * 2);
  __bf16* WqT    = (__bf16*)carve((size_t)DM * DM * 2);
  __bf16* WkT    = (__bf16*)carve((size_t)DM * DM * 2);
  __bf16* WvT    = (__bf16*)carve((size_t)DM * DM * 2);
  __bf16* WoT    = (__bf16*)carve((size_t)DM * DM * 2);
  __bf16* W1T    = (__bf16*)carve((size_t)DM * DFF * 2);
  __bf16* W2T    = (__bf16*)carve((size_t)DFF * DM * 2);
  __bf16* Qbf    = (__bf16*)carve((size_t)MTOT * DM * 2);
  __bf16* Ktbf   = (__bf16*)carve((size_t)DM * MTOT * 2);
  __bf16* Vbf    = (__bf16*)carve((size_t)MTOT * DM * 2);
  __bf16* ctxbf  = (__bf16*)carve((size_t)MTOT * DM * 2);
  float*  x1     = (float*)carve((size_t)MTOT * DM * 4);
  __bf16* h2_bf  = (__bf16*)carve((size_t)MTOT * DM * 2);
  __bf16* ff1_bf = (__bf16*)carve((size_t)MTOT * DFF * 2);

  // 1) LN1
  ln_bf16_kernel<<<MTOT, 256, 0, stream>>>(x, g1, be1, h_bf);

  // 2) weight transposes (f32 -> bf16, [out][in] -> [in][out])
  {
    int n1 = DM * DM, n2 = DFF * DM;
    transpose_bf16_kernel<<<(n1 + 255) / 256, 256, 0, stream>>>(Wq, WqT, DM, DM);
    transpose_bf16_kernel<<<(n1 + 255) / 256, 256, 0, stream>>>(Wk, WkT, DM, DM);
    transpose_bf16_kernel<<<(n1 + 255) / 256, 256, 0, stream>>>(Wv, WvT, DM, DM);
    transpose_bf16_kernel<<<(n1 + 255) / 256, 256, 0, stream>>>(Wo, WoT, DM, DM);
    transpose_bf16_kernel<<<(n2 + 255) / 256, 256, 0, stream>>>(W1, W1T, DFF, DM);
    transpose_bf16_kernel<<<(n2 + 255) / 256, 256, 0, stream>>>(W2, W2T, DM, DFF);
  }

  // 3) QKV projections (block tile 128x128)
  dim3 gQ(DM / 128, MTOT / 128);
  gemm_wmma_kernel<0><<<gQ, 256, 0, stream>>>(h_bf, WqT, bq, nullptr, Qbf, MTOT, DM, DM);
  gemm_wmma_kernel<1><<<gQ, 256, 0, stream>>>(h_bf, WkT, bk, nullptr, Ktbf, MTOT, DM, DM);
  gemm_wmma_kernel<0><<<gQ, 256, 0, stream>>>(h_bf, WvT, bv, nullptr, Vbf, MTOT, DM, DM);

  // 4) attention (2 batches * 16 heads * 128 q-tiles = 4096 wave-tiles, 4/block)
  flash_attn_kernel<<<1024, 128, 0, stream>>>(Qbf, Ktbf, Vbf, ctxbf);

  // 5) output projection + residual -> x1 (f32)
  gemm_wmma_kernel<2><<<gQ, 256, 0, stream>>>(ctxbf, WoT, bo, x, x1, MTOT, DM, DM);

  // 6) LN2
  ln_bf16_kernel<<<MTOT, 256, 0, stream>>>(x1, g2, be2, h2_bf);

  // 7) FFN
  dim3 gF1(DFF / 128, MTOT / 128);
  gemm_wmma_kernel<3><<<gF1, 256, 0, stream>>>(h2_bf, W1T, b1, nullptr, ff1_bf, MTOT, DFF, DM);
  gemm_wmma_kernel<2><<<gQ, 256, 0, stream>>>(ff1_bf, W2T, b2, x1, out, MTOT, DM, DFF);
}
